// NunchakuSanaTransformerBlocks_79817672229295
// MI455X (gfx1250) — matrix operand order repaired and checked by reading, not compiled, not run on hardware
//
#include <hip/hip_runtime.h>

#define DEV __device__ __forceinline__

typedef __attribute__((ext_vector_type(16))) __bf16 v16bf;
typedef __attribute__((ext_vector_type(8)))  float  v8f;

struct u32x8 { uint4 lo, hi; };

DEV unsigned short f2us(float f) {               // hardware f32->bf16 (RNE)
  return __builtin_bit_cast(unsigned short, (__bf16)f);
}
DEV unsigned packbf(float a, float b) {          // lo=bf16(a), hi=bf16(b): 1 instr
  unsigned r;
  asm("v_cvt_pk_bf16_f32 %0, %1, %2" : "=v"(r) : "v"(a), "v"(b));
  return r;
}
DEV float us2f(unsigned short h) {
  unsigned u = ((unsigned)h) << 16;
  return __builtin_bit_cast(float, u);
}

// ---- problem dims (Sana-1.6B block) ----
constexpr int B_   = 2;
constexpr int NI   = 1024;
constexpr int T_   = 300;
constexpr int D_   = 2240;
constexpr int H_   = 70;
constexpr int DFF_ = 5600;
constexpr int ROWS  = B_ * NI;   // 2048
constexpr int TROWS = B_ * T_;   // 600
constexpr int D6    = 6 * D_;    // 13440

// ---- workspace layout (bytes) ----
constexpr size_t WS_MOD  = 0;                                   // B*6*D f32
constexpr size_t WS_HMOD = 131072;                              // 2048*2240 bf16 (hmod / f)
constexpr size_t WS_BIG  = WS_HMOD + (size_t)ROWS * D_ * 2;     // qkv f32 (later: u bf16)
constexpr size_t WS_OBUF = WS_BIG  + (size_t)ROWS * 3 * D_ * 4; // o_self / o_cross bf16
constexpr size_t WS_X1   = WS_OBUF + (size_t)ROWS * D_ * 2;     // x1 f32
constexpr size_t WS_XB   = WS_X1   + (size_t)ROWS * D_ * 4;     // x1 bf16
constexpr size_t WS_X2   = WS_XB   + (size_t)ROWS * D_ * 2;     // x2 f32
constexpr size_t WS_Q    = WS_X2   + (size_t)ROWS * D_ * 4;     // q_cross f32
constexpr size_t WS_KVC  = WS_Q    + (size_t)ROWS * D_ * 4;     // kvc f32 (600 x 4480)
constexpr size_t WS_HG   = WS_KVC  + (size_t)TROWS * 2 * D_ * 4;// hgate bf16 (2048 x 5600)
constexpr size_t WS_LA   = WS_HG   + (size_t)ROWS * DFF_ * 2;   // per-(b,h) kv(32x32)+ksum(32)
constexpr size_t WS_ENC  = WS_LA   + (size_t)B_ * H_ * 1056 * 4;// enc bf16 (600 x 2240)

// ============================================================
// adaLN-single modulation: mod[b,i,j] = (temb @ W_t + b_t) + sst
// ============================================================
__global__ __launch_bounds__(256) void k_mod(const int* __restrict__ ts,
                                             const float* __restrict__ Wt,
                                             const float* __restrict__ bt,
                                             const float* __restrict__ sst,
                                             float* __restrict__ mod) {
  __shared__ float temb[256];
  const int b = blockIdx.x, tid = threadIdx.x;
  if (tid < 128) {
    float f = __expf(-9.210340371976184f * (float)tid * (1.0f / 128.0f));
    float ang = (float)ts[b] * f;
    temb[tid]       = __cosf(ang);
    temb[tid + 128] = __sinf(ang);
  }
  __syncthreads();
  for (int j = tid; j < D6; j += 256) {
    float acc = bt[j] + sst[j];
    for (int k = 0; k < 256; ++k) acc += temb[k] * Wt[(size_t)k * D6 + j];
    mod[(size_t)b * D6 + j] = acc;
  }
}

// ============================================================
// LN (no affine) * (1+sc) + sh  -> bf16
// ============================================================
__global__ __launch_bounds__(256) void k_ln_mod(const float* __restrict__ x,
                                                const float* __restrict__ mod,
                                                int shIdx, int scIdx,
                                                unsigned short* __restrict__ out) {
  __shared__ float red[256];
  const int r = blockIdx.x, tid = threadIdx.x, b = r / NI;
  const float* xr = x + (size_t)r * D_;
  float lv[9]; int n = 0;
  float s = 0.f;
  for (int j = tid; j < D_; j += 256) { float v = xr[j]; lv[n++] = v; s += v; }
  red[tid] = s; __syncthreads();
  for (int off = 128; off > 0; off >>= 1) { if (tid < off) red[tid] += red[tid + off]; __syncthreads(); }
  const float mean = red[0] * (1.0f / D_);
  __syncthreads();
  float s2 = 0.f;
  for (int i = 0; i < n; ++i) { float d = lv[i] - mean; s2 += d * d; }
  red[tid] = s2; __syncthreads();
  for (int off = 128; off > 0; off >>= 1) { if (tid < off) red[tid] += red[tid + off]; __syncthreads(); }
  const float rs = rsqrtf(red[0] * (1.0f / D_) + 1e-6f);
  const float* sh = mod + ((size_t)b * 6 + shIdx) * D_;
  const float* sc = mod + ((size_t)b * 6 + scIdx) * D_;
  n = 0;
  for (int j = tid; j < D_; j += 256) {
    float v = (lv[n++] - mean) * rs * (1.0f + sc[j]) + sh[j];
    out[(size_t)r * D_ + j] = f2us(v);
  }
}

// ============================================================
// f32 -> bf16 convert
// ============================================================
__global__ __launch_bounds__(256) void k_f2bf(const float* __restrict__ in,
                                              unsigned short* __restrict__ out, int n) {
  int i = blockIdx.x * 256 + threadIdx.x;
  if (i < n) out[i] = f2us(in[i]);
}

// ============================================================
// Generic GEMM: C[M,N] = A_bf16[M,K] @ W_f32[K,N], epilogue fused.
// Block 256 thr (8 waves); tile 128x64, K-step 32 == one WMMA chunk.
// Wave grid 4(M)x2(N); each wave: 2 M-tiles x 2 N-tiles -> 4 v_wmma per step.
// Double-buffered LDS weight slab -> single barrier per K-step.
// GUARD=false: M is a multiple of 128 (no bounds logic in hot path).
// ============================================================
template <bool GUARD>
__global__ __launch_bounds__(256) void gemm_bf16_wmma(
    const unsigned short* __restrict__ A, const float* __restrict__ W,
    int M, int N, int K,
    float* __restrict__ outF, unsigned short* __restrict__ outB,
    const float* __restrict__ resid, const float* __restrict__ gate,
    int gateStride, int rowsPerBatch) {
  __shared__ __align__(16) unsigned short Bs[2][64][40];  // [buf][n][k], padded stride
  const int tid  = threadIdx.x;
  const int lane = tid & 31;
  const int wid  = tid >> 5;
  const int rowBase = blockIdx.y * 128;
  const int colBase = blockIdx.x * 64;
  const int wm = wid & 3;          // 4 M-groups of 32 rows
  const int wn = wid >> 2;         // 2 N-groups of 32 cols
  const int nIdx  = lane & 15;
  const int kofsA = (lane & 16) ? 16 : 0;   // bf16 A-frag K-offset per half-wave
  const int kofsB = (lane & 16) ? 16 : 0;   // bf16 B-frag K-offset per half-wave

  // A fragment row pointers (two 16-row tiles per wave), bumped by 32/step
  const int aRow0 = rowBase + wm * 32 + nIdx;
  const int aRow1 = aRow0 + 16;
  const bool av0 = !GUARD || (aRow0 < M);
  const bool av1 = !GUARD || (aRow1 < M);
  const unsigned short* ap0 = A + (size_t)aRow0 * K + kofsA;
  const unsigned short* ap1 = A + (size_t)aRow1 * K + kofsA;

  // B staging: thread -> (kk = 2*(tid>>4) [two adjacent k-rows], j0 = (tid&15)*4)
  const int skk = (tid >> 4) * 2;
  const int sj  = (tid & 15) * 4;
  const float* wp0 = W + (size_t)skk * N + colBase + sj;   // row kk
  const float* wp1 = wp0 + N;                              // row kk+1
  unsigned short* bsl = &Bs[0][0][0];
  const int sIdx = sj * 40 + skk;                          // ushort index, kk even
  const size_t wstep = (size_t)32 * N;

  // B fragment LDS offsets (constant across K loop)
  const int f0 = (wn * 32 + nIdx) * 40 + kofsB;
  const int f1 = f0 + 16 * 40;

  v8f acc00 = {0.f,0.f,0.f,0.f,0.f,0.f,0.f,0.f};
  v8f acc01 = acc00, acc10 = acc00, acc11 = acc00;

  int p = 0;
  for (int kb = 0; kb < K; kb += 32) {
    // ---- stage + convert 32x64 weight slab: 2 b128 loads, 4 cvt_pk, 4 b32 LDS stores ----
    unsigned short* buf = bsl + p * (64 * 40);
    float4 w0 = *(const float4*)wp0;
    float4 w1 = *(const float4*)wp1;
    *(unsigned*)(buf + sIdx)            = packbf(w0.x, w1.x);
    *(unsigned*)(buf + sIdx + 40)       = packbf(w0.y, w1.y);
    *(unsigned*)(buf + sIdx + 80)       = packbf(w0.z, w1.z);
    *(unsigned*)(buf + sIdx + 120)      = packbf(w0.w, w1.w);
    wp0 += wstep; wp1 += wstep;
    if (kb + 32 < K) __builtin_prefetch(wp0, 0, 3);
    __syncthreads();

    // ---- A fragments: two aligned b128 global loads per lane per tile ----
    u32x8 at0, at1;
    if (av0) { at0.lo = *(const uint4*)ap0; at0.hi = *(const uint4*)(ap0 + 8); }
    else     { at0.lo = make_uint4(0u,0u,0u,0u); at0.hi = make_uint4(0u,0u,0u,0u); }
    if (av1) { at1.lo = *(const uint4*)ap1; at1.hi = *(const uint4*)(ap1 + 8); }
    else     { at1.lo = make_uint4(0u,0u,0u,0u); at1.hi = make_uint4(0u,0u,0u,0u); }
    ap0 += 32; ap1 += 32;
    v16bf a0 = __builtin_bit_cast(v16bf, at0);
    v16bf a1 = __builtin_bit_cast(v16bf, at1);

    // ---- B fragments: contiguous 32B LDS reads ----
    v16bf b0 = __builtin_bit_cast(v16bf, *(const u32x8*)(buf + f0));
    v16bf b1 = __builtin_bit_cast(v16bf, *(const u32x8*)(buf + f1));

    acc00 = __builtin_amdgcn_wmma_f32_16x16x32_bf16(false, a0, false, b0, (short)0, acc00, false, false);
    acc01 = __builtin_amdgcn_wmma_f32_16x16x32_bf16(false, a0, false, b1, (short)0, acc01, false, false);
    acc10 = __builtin_amdgcn_wmma_f32_16x16x32_bf16(false, a1, false, b0, (short)0, acc10, false, false);
    acc11 = __builtin_amdgcn_wmma_f32_16x16x32_bf16(false, a1, false, b1, (short)0, acc11, false, false);
    p ^= 1;     // double buffer: no trailing barrier needed
  }

  // ---- epilogue: C/D layout -> row = r + (lane>=16 ? 8 : 0), col = lane%16 ----
  const int col0 = colBase + wn * 32 + nIdx;
  const int col1 = col0 + 16;
#pragma unroll
  for (int mi = 0; mi < 2; ++mi) {
    v8f a = mi ? acc10 : acc00;
    v8f b = mi ? acc11 : acc01;
    const int mBase = rowBase + wm * 32 + mi * 16 + ((lane & 16) ? 8 : 0);
#pragma unroll
    for (int r = 0; r < 8; ++r) {
      int row = mBase + r;
      if (!GUARD || row < M) {
        float v0 = a[r], v1 = b[r];
        if (gate) {
          int bb = row / rowsPerBatch;
          v0 = resid[(size_t)row * N + col0] + gate[(size_t)bb * gateStride + col0] * v0;
          v1 = resid[(size_t)row * N + col1] + gate[(size_t)bb * gateStride + col1] * v1;
        } else if (resid) {
          v0 += resid[(size_t)row * N + col0];
          v1 += resid[(size_t)row * N + col1];
        }
        if (outF) {
          outF[(size_t)row * N + col0] = v0;
          outF[(size_t)row * N + col1] = v1;
        }
        if (outB) {
          outB[(size_t)row * N + col0] = f2us(v0);
          outB[(size_t)row * N + col1] = f2us(v1);
        }
      }
    }
  }
}

// ============================================================
// ReLU linear attention, stage 1: kv[d,e] = sum_n k_d v_e ; ksum
// ============================================================
__global__ __launch_bounds__(256) void k_lin_kv(const float* __restrict__ qkv,
                                                float* __restrict__ la) {
  __shared__ float kb[8][32], vb[8][32], ks[32];
  const int bh = blockIdx.x, b = bh / H_, h = bh % H_, tid = threadIdx.x;
  const int d = tid >> 3, e0 = (tid & 7) * 4;
  float acc[4] = {0.f, 0.f, 0.f, 0.f};
  if (tid < 32) ks[tid] = 0.f;
  for (int n0 = 0; n0 < NI; n0 += 8) {
    int rr = tid >> 5, dd = tid & 31;
    size_t base = (size_t)(b * NI + n0 + rr) * (3 * D_) + h * 32 + dd;
    kb[rr][dd] = fmaxf(qkv[base + D_], 0.f);
    vb[rr][dd] = qkv[base + 2 * D_];
    __syncthreads();
#pragma unroll
    for (int i = 0; i < 8; ++i) {
      float kd = kb[i][d];
#pragma unroll
      for (int j = 0; j < 4; ++j) acc[j] += kd * vb[i][e0 + j];
    }
    if (tid < 32) {
#pragma unroll
      for (int i = 0; i < 8; ++i) ks[tid] += kb[i][tid];
    }
    __syncthreads();
  }
  float* o = la + (size_t)bh * 1056;
#pragma unroll
  for (int j = 0; j < 4; ++j) o[d * 32 + e0 + j] = acc[j];
  if (tid < 32) o[1024 + tid] = ks[tid];
}

// ============================================================
// ReLU linear attention, stage 2: o = (q . kv) / (q . ksum + 1e-6)
// ============================================================
__global__ __launch_bounds__(256) void k_lin_out(const float* __restrict__ qkv,
                                                 const float* __restrict__ la,
                                                 unsigned short* __restrict__ oself) {
  __shared__ float kvs[1024], kss[32], qs[8][32];
  const int bh = blockIdx.y, b = bh / H_, h = bh % H_;
  const int n0 = blockIdx.x * 8, tid = threadIdx.x;
  const float* base = la + (size_t)bh * 1056;
#pragma unroll
  for (int i = 0; i < 4; ++i) kvs[tid + 256 * i] = base[tid + 256 * i];
  if (tid < 32) kss[tid] = base[1024 + tid];
  {
    int rr = tid >> 5, dd = tid & 31;
    qs[rr][dd] = fmaxf(qkv[(size_t)(b * NI + n0 + rr) * (3 * D_) + h * 32 + dd], 0.f);
  }
  __syncthreads();
  const int nl = tid >> 5, e = tid & 31;
  float z = 1e-6f, o = 0.f;
#pragma unroll
  for (int d = 0; d < 32; ++d) {
    float qd = qs[nl][d];
    z += qd * kss[d];
    o += qd * kvs[d * 32 + e];
  }
  oself[(size_t)(b * NI + n0 + nl) * D_ + h * 32 + e] = f2us(o / z);
}

// ============================================================
// Masked cross attention (varlen via additive mask), wave-per-row softmax.
// Block: (b,h, 64 image tokens). kc/vc tile staged in LDS as bf16.
// ============================================================
__global__ __launch_bounds__(256) void k_cross(const float* __restrict__ qc,
                                               const float* __restrict__ kvc,
                                               const float* __restrict__ maskp,
                                               unsigned short* __restrict__ ocross) {
  __shared__ unsigned short kcs[T_ * 32];
  __shared__ unsigned short vcs[T_ * 32];
  __shared__ float qsf[64 * 32];
  __shared__ float pb[8][304];
  __shared__ float msk[T_];
  const int bh = blockIdx.y, b = bh / H_, h = bh % H_;
  const int n0g = blockIdx.x * 64, tid = threadIdx.x;

  for (int idx = tid; idx < T_ * 32; idx += 256) {
    int t = idx >> 5, d = idx & 31;
    size_t base = (size_t)(b * T_ + t) * (2 * D_) + h * 32 + d;
    kcs[idx] = f2us(kvc[base]);
    vcs[idx] = f2us(kvc[base + D_]);
  }
  for (int idx = tid; idx < 64 * 32; idx += 256) {
    int r = idx >> 5, d = idx & 31;
    qsf[idx] = qc[(size_t)(b * NI + n0g + r) * D_ + h * 32 + d];
  }
  for (int idx = tid; idx < T_; idx += 256) msk[idx] = maskp[b * T_ + idx];
  __syncthreads();

  const int w = tid >> 5, lane = tid & 31;
  const float inv_sqrt_dh = 0.17677669529663688f;   // 1/sqrt(32)
  for (int rr = 0; rr < 8; ++rr) {
    const int row = w * 8 + rr;
    const float* qrow = qsf + row * 32;
    float sv[10];
    int cnt = 0;
    float mx = -1e30f;
    for (int t = lane; t < T_; t += 32) {
      float s = 0.f;
#pragma unroll
      for (int d = 0; d < 32; ++d) s += qrow[d] * us2f(kcs[t * 32 + d]);
      s = s * inv_sqrt_dh + msk[t];
      sv[cnt++] = s;
      mx = fmaxf(mx, s);
    }
#pragma unroll
    for (int off = 16; off > 0; off >>= 1) mx = fmaxf(mx, __shfl_xor(mx, off, 32));
    float sum = 0.f;
    cnt = 0;
    for (int t = lane; t < T_; t += 32) {
      float p = __expf(sv[cnt++] - mx);
      pb[w][t] = p;
      sum += p;
    }
#pragma unroll
    for (int off = 16; off > 0; off >>= 1) sum += __shfl_xor(sum, off, 32);
    __syncthreads();
    float acc = 0.f;
    for (int t = 0; t < T_; ++t) acc += pb[w][t] * us2f(vcs[t * 32 + lane]);
    ocross[(size_t)(b * NI + n0g + row) * D_ + h * 32 + lane] = f2us(acc / sum);
    __syncthreads();
  }
}

// ============================================================
// SwiGLU: h = silu(u[:, :DFF]) * u[:, DFF:]
// ============================================================
__global__ __launch_bounds__(256) void k_swiglu(const unsigned short* __restrict__ u,
                                                unsigned short* __restrict__ hg) {
  int idx = blockIdx.x * 256 + threadIdx.x;
  if (idx >= ROWS * DFF_) return;
  int r = idx / DFF_, j = idx - r * DFF_;
  float a = us2f(u[(size_t)r * (2 * DFF_) + j]);
  float g = us2f(u[(size_t)r * (2 * DFF_) + DFF_ + j]);
  float s = a * (1.0f / (1.0f + __expf(-a)));
  hg[idx] = f2us(s * g);
}

// ============================================================
extern "C" void kernel_launch(void* const* d_in, const int* in_sizes, int n_in,
                              void* d_out, int out_size, void* d_ws, size_t ws_size,
                              hipStream_t stream) {
  const float* x_in  = (const float*)d_in[0];
  const float* enc   = (const float*)d_in[1];
  const float* maskp = (const float*)d_in[2];
  const int*   ts    = (const int*)d_in[3];
  const float* W_t   = (const float*)d_in[4];
  const float* b_t   = (const float*)d_in[5];
  const float* sst   = (const float*)d_in[6];
  const float* W_qkv = (const float*)d_in[7];
  const float* W_o   = (const float*)d_in[8];
  const float* W_q   = (const float*)d_in[9];
  const float* W_kv  = (const float*)d_in[10];
  const float* W_co  = (const float*)d_in[11];
  const float* W_ff1 = (const float*)d_in[12];
  const float* W_ff2 = (const float*)d_in[13];
  float* out = (float*)d_out;

  char* ws = (char*)d_ws;
  float*          mod   = (float*)(ws + WS_MOD);
  unsigned short* hmod  = (unsigned short*)(ws + WS_HMOD);
  float*          qkv   = (float*)(ws + WS_BIG);
  unsigned short* ubuf  = (unsigned short*)(ws + WS_BIG);
  unsigned short* obuf  = (unsigned short*)(ws + WS_OBUF);
  float*          x1    = (float*)(ws + WS_X1);
  unsigned short* xb    = (unsigned short*)(ws + WS_XB);
  float*          x2    = (float*)(ws + WS_X2);
  float*          qc    = (float*)(ws + WS_Q);
  float*          kvc   = (float*)(ws + WS_KVC);
  unsigned short* hg    = (unsigned short*)(ws + WS_HG);
  float*          la    = (float*)(ws + WS_LA);
  unsigned short* encb  = (unsigned short*)(ws + WS_ENC);

  const int gy  = ROWS / 128;            // 16 (exact -> GUARD=false)
  const int gyT = (TROWS + 127) / 128;   // 5  (ragged -> GUARD=true)

  // adaLN modulation table
  k_mod<<<B_, 256, 0, stream>>>(ts, W_t, b_t, sst, mod);

  // ---- self attention (ReLU linear attention) ----
  k_ln_mod<<<ROWS, 256, 0, stream>>>(x_in, mod, 0, 1, hmod);                 // sh1, sc1
  gemm_bf16_wmma<false><<<dim3(3 * D_ / 64, gy), 256, 0, stream>>>(
      hmod, W_qkv, ROWS, 3 * D_, D_, qkv, nullptr, nullptr, nullptr, 0, NI);
  k_lin_kv<<<B_ * H_, 256, 0, stream>>>(qkv, la);
  k_lin_out<<<dim3(NI / 8, B_ * H_), 256, 0, stream>>>(qkv, la, obuf);
  gemm_bf16_wmma<false><<<dim3(D_ / 64, gy), 256, 0, stream>>>(
      obuf, W_o, ROWS, D_, D_, x1, xb, x_in, mod + 2 * D_, D6, NI);          // x1 = x + g1*o@Wo

  // ---- cross attention to text tokens ----
  gemm_bf16_wmma<false><<<dim3(D_ / 64, gy), 256, 0, stream>>>(
      xb, W_q, ROWS, D_, D_, qc, nullptr, nullptr, nullptr, 0, NI);
  k_f2bf<<<(TROWS * D_ + 255) / 256, 256, 0, stream>>>(enc, encb, TROWS * D_);
  gemm_bf16_wmma<true><<<dim3(2 * D_ / 64, gyT), 256, 0, stream>>>(
      encb, W_kv, TROWS, 2 * D_, D_, kvc, nullptr, nullptr, nullptr, 0, NI);
  k_cross<<<dim3(NI / 64, B_ * H_), 256, 0, stream>>>(qc, kvc, maskp, obuf);
  gemm_bf16_wmma<false><<<dim3(D_ / 64, gy), 256, 0, stream>>>(
      obuf, W_co, ROWS, D_, D_, x2, nullptr, x1, nullptr, 0, NI);            // x2 = x1 + o@Wco

  // ---- gated FFN ----
  k_ln_mod<<<ROWS, 256, 0, stream>>>(x2, mod, 3, 4, hmod);                   // sh2, sc2
  gemm_bf16_wmma<false><<<dim3(2 * DFF_ / 64, gy), 256, 0, stream>>>(
      hmod, W_ff1, ROWS, 2 * DFF_, D_, nullptr, ubuf, nullptr, nullptr, 0, NI);
  k_swiglu<<<(ROWS * DFF_ + 255) / 256, 256, 0, stream>>>(ubuf, hg);
  gemm_bf16_wmma<false><<<dim3(D_ / 64, gy), 256, 0, stream>>>(
      hg, W_ff2, ROWS, D_, DFF_, out, nullptr, x2, mod + 5 * D_, D6, NI);    // out = x2 + g2*h@Wff2
}
